// SingleHeadAttention_84688165143254
// MI455X (gfx1250) — matrix operand order repaired and checked by reading, not compiled, not run on hardware
//
#include <hip/hip_runtime.h>
#include <hip/hip_bf16.h>
#include <cstddef>

// ---------------------------------------------------------------------------
// Single-head causal attention, B=8 S=2048 E=768 D=64, f32 in/out.
// bf16 WMMA (v_wmma_f32_16x16x32_bf16) for QKV projection, QK^T and PV.
// Workspace layout (bytes):
//   [0,            294912)   : Wpack  (3 x 24 x 4 fragments x 1KB, bf16)
//   [294912,      2392064)   : Qb     (16384 x 64 bf16, row major)
//   [2392064,     4489216)   : Kb     (16384 x 64 bf16, row major)
//   [4489216,     6586368)   : VT     (64 x 16384 bf16, V transposed)
// ---------------------------------------------------------------------------

#define SEQ     2048
#define NTOK    16384   // 8 * 2048
#define EMB     768
#define HD      64

typedef __attribute__((ext_vector_type(16))) __bf16        v16bf;
typedef __attribute__((ext_vector_type(8)))  float         v8f;
typedef __attribute__((ext_vector_type(8)))  unsigned int  v8u;
typedef __attribute__((ext_vector_type(4)))  unsigned int  v4u;
typedef __attribute__((ext_vector_type(4)))  float         v4f;

__device__ __forceinline__ unsigned short f2bf(float x) {
  unsigned int u = __builtin_bit_cast(unsigned int, x);
  u = (u + 0x7FFFu + ((u >> 16) & 1u)) >> 16;   // round-to-nearest-even
  return (unsigned short)u;
}
__device__ __forceinline__ unsigned int pk2(float lo, float hi) {
  return (unsigned int)f2bf(lo) | ((unsigned int)f2bf(hi) << 16);
}
__device__ __forceinline__ v16bf mk16(v4u c0, v4u c1) {
  v8u u;
  u[0] = c0[0]; u[1] = c0[1]; u[2] = c0[2]; u[3] = c0[3];
  u[4] = c1[0]; u[5] = c1[1]; u[6] = c1[2]; u[7] = c1[3];
  return __builtin_bit_cast(v16bf, u);
}
// load 16 consecutive bf16 (32B, 32B-aligned) -> B/A fragment half
__device__ __forceinline__ v16bf ldb32(const unsigned short* p) {
  v4u c0 = *(const v4u*)p;
  v4u c1 = *(const v4u*)(p + 8);
  return mk16(c0, c1);
}
__device__ __forceinline__ v8f wmma_bf16(v16bf a, v16bf b, v8f c) {
  return __builtin_amdgcn_wmma_f32_16x16x32_bf16(
      /*neg_a=*/false, a, /*neg_b=*/false, b,
      /*c_mod=*/(short)0, c, /*reuse_a=*/false, /*reuse_b=*/false);
}
// async global -> LDS copy, 16B per lane (tracked by ASYNCcnt)
__device__ __forceinline__ void async_g2l_b128(unsigned lds_off, const void* g) {
  unsigned long long ga = (unsigned long long)g;
  asm volatile("global_load_async_to_lds_b128 %0, %1, off"
               :: "v"(lds_off), "v"(ga) : "memory");
}

// ---------------------------------------------------------------------------
// Kernel 1: pack Wq/Wk/Wv (f32 [768x64] row-major) into bf16 B-fragment tiles.
// Fragment (kt,nt) covers K=kt*32..+31, N=nt*16..+15; per lane 16 bf16:
//   col n = nt*16 + (lane&15);  k = kt*32 + (lane>>4)*16 + j,  j=0..15
// Fragment index: (w*96 + kt*4 + nt), each fragment 512 bf16 (1KB).
// ---------------------------------------------------------------------------
__global__ void __launch_bounds__(32)
pack_weights(const float* __restrict__ Wq, const float* __restrict__ Wk,
             const float* __restrict__ Wv, unsigned short* __restrict__ Wpack) {
  const int frag = blockIdx.x;          // 0..287
  const int lane = threadIdx.x & 31;
  const int w  = frag / 96;
  const int kt = (frag % 96) >> 2;
  const int nt = frag & 3;
  const float* W = (w == 0) ? Wq : ((w == 1) ? Wk : Wv);
  const int col   = nt * 16 + (lane & 15);
  const int kbase = kt * 32 + (lane >> 4) * 16;
  unsigned short* dst = Wpack + (size_t)frag * 512 + lane * 16;
#pragma unroll
  for (int j = 0; j < 16; ++j)
    dst[j] = f2bf(W[(size_t)(kbase + j) * HD + col]);
}

// ---------------------------------------------------------------------------
// Kernel 2: QKV projection. 8 waves/block, one 16-token tile per wave.
// Weights staged through a double-buffered LDS slice (12 frags = 12KB per kt
// step, 3 x 16B chunks per thread) via GLOBAL_LOAD_ASYNC_TO_LDS_B128; the
// async copy of slice kt+1 overlaps the 12 WMMAs of slice kt (B operands via
// ds_load_b128). Q,K stored row-major bf16; V transposed (VT[d][tok]).
// ---------------------------------------------------------------------------
__global__ void __launch_bounds__(256)
qkv_kernel(const float* __restrict__ X, const unsigned short* __restrict__ Wpack,
           const float* __restrict__ bq, const float* __restrict__ bk,
           const float* __restrict__ bv,
           unsigned short* __restrict__ Qb, unsigned short* __restrict__ Kb,
           unsigned short* __restrict__ VT) {
  __shared__ unsigned short Wlds[2][12 * 512];   // 2 x 12KB

  const int tid  = threadIdx.x;
  const int lane = tid & 31;
  const int ln   = lane & 15;
  const int half = lane >> 4;
  const int tok0 = (blockIdx.x * 8 + (tid >> 5)) * 16;

  // slice kt = 3 w-regions x 4 frags x 1KB = 12KB = 768 x 16B chunks;
  // 3 chunks per thread, coalesced (idx = j*256 + tid).
  const unsigned ldsBase = (unsigned)(size_t)(&Wlds[0][0]);  // low 32b = LDS offset

#pragma unroll
  for (int j = 0; j < 3; ++j) {
    const int idx = j * 256 + tid;                  // 0..767
    const int w = idx >> 8, rem = idx & 255;
    const unsigned short* g =
        Wpack + (size_t)(w * 96 + 0 * 4) * 512 + (size_t)rem * 8;
    async_g2l_b128(ldsBase + (unsigned)idx * 16, g);
  }
  asm volatile("s_wait_asynccnt 0x0" ::: "memory");
  __syncthreads();

  v8f acc[3][4];
#pragma unroll
  for (int w = 0; w < 3; ++w)
#pragma unroll
    for (int nt = 0; nt < 4; ++nt) acc[w][nt] = (v8f)0.f;

  for (int kt = 0; kt < 24; ++kt) {
    // kick off async copy of slice kt+1 into the other LDS buffer
    if (kt < 23) {
      const unsigned dst = ldsBase + ((kt + 1) & 1) * (12 * 512 * 2);
#pragma unroll
      for (int j = 0; j < 3; ++j) {
        const int idx = j * 256 + tid;
        const int w = idx >> 8, rem = idx & 255;
        const unsigned short* g =
            Wpack + (size_t)(w * 96 + (kt + 1) * 4) * 512 + (size_t)rem * 8;
        async_g2l_b128(dst + (unsigned)idx * 16, g);
      }
    }

    // A fragment from f32 X: row = ln, K = half*16+0..15
    const float* xp = X + (size_t)(tok0 + ln) * EMB + kt * 32 + half * 16;
    v4f x0 = *(const v4f*)(xp);
    v4f x1 = *(const v4f*)(xp + 4);
    v4f x2 = *(const v4f*)(xp + 8);
    v4f x3 = *(const v4f*)(xp + 12);
    v8u au;
    au[0] = pk2(x0[0], x0[1]); au[1] = pk2(x0[2], x0[3]);
    au[2] = pk2(x1[0], x1[1]); au[3] = pk2(x1[2], x1[3]);
    au[4] = pk2(x2[0], x2[1]); au[5] = pk2(x2[2], x2[3]);
    au[6] = pk2(x3[0], x3[1]); au[7] = pk2(x3[2], x3[3]);
    v16bf a = __builtin_bit_cast(v16bf, au);

    // 12 WMMAs, B operands from LDS (ds_load_b128 pairs)
    const unsigned short* bs = &Wlds[kt & 1][0] + lane * 16;
#pragma unroll
    for (int w = 0; w < 3; ++w)
#pragma unroll
      for (int nt = 0; nt < 4; ++nt)
        acc[w][nt] = wmma_bf16(a, ldb32(bs + (size_t)(w * 4 + nt) * 512),
                               acc[w][nt]);

    // drain this wave's async writes, then publish to the workgroup
    if (kt < 23) asm volatile("s_wait_asynccnt 0x0" ::: "memory");
    __syncthreads();
  }

  // bias + stores (C layout: row = v + 8*half, col = ln within n-tile)
#pragma unroll
  for (int nt = 0; nt < 4; ++nt) {
    const int col = nt * 16 + ln;
    const float bQ = bq[col], bK = bk[col], bV = bv[col];
#pragma unroll
    for (int v = 0; v < 8; ++v) {
      const int tok = tok0 + v + 8 * half;
      Qb[(size_t)tok * HD + col] = f2bf(acc[0][nt][v] + bQ);
      Kb[(size_t)tok * HD + col] = f2bf(acc[1][nt][v] + bK);
      VT[(size_t)col * NTOK + tok] = f2bf(acc[2][nt][v] + bV);
    }
  }
}

// ---------------------------------------------------------------------------
// Kernel 3: flash attention. One wave per 16-query tile; 32 keys per step.
// ---------------------------------------------------------------------------
__global__ void __launch_bounds__(32)
flash_kernel(const unsigned short* __restrict__ Qb,
             const unsigned short* __restrict__ Kb,
             const unsigned short* __restrict__ VT,
             float* __restrict__ out) {
  __shared__ unsigned short Pt[16 * 32];   // P tile staging (bf16, row-major)

  const int wid  = blockIdx.x;             // 0..1023
  const int lane = threadIdx.x & 31;
  const int ln   = lane & 15;
  const int half = lane >> 4;
  const int bIdx = wid >> 7;               // batch
  const int q0   = (wid & 127) << 4;       // query offset within batch
  const int bTok0 = bIdx * SEQ;
  const int tokQ0 = bTok0 + q0;

  // Q A-fragments for d = 0..31 and 32..63
  v16bf qa[2];
#pragma unroll
  for (int h = 0; h < 2; ++h) {
    const unsigned short* qp = Qb + (size_t)(tokQ0 + ln) * HD + h * 32 + half * 16;
    qa[h] = mk16(*(const v4u*)qp, *(const v4u*)(qp + 8));
  }

  v8f O[4];
#pragma unroll
  for (int nt = 0; nt < 4; ++nt) O[nt] = (v8f)0.f;
  float m[8], l[8];
#pragma unroll
  for (int v = 0; v < 8; ++v) { m[v] = -1e30f; l[v] = 0.f; }

  for (int j0 = 0; j0 < q0 + 16; j0 += 32) {
    // prefetch next key block into caches (global_prefetch_b8)
    if (j0 + 32 < q0 + 16) {
      __builtin_prefetch(Kb + (size_t)(bTok0 + j0 + 32 + ln) * HD, 0, 1);
      __builtin_prefetch(Kb + (size_t)(bTok0 + j0 + 48 + ln) * HD, 0, 1);
    }

    // ---- S = Q K^T (16 x 32): hoist all 4 K fragments, then 4 WMMAs
    const int kj0 = j0 + ln;
    const int kj1 = j0 + 16 + ln;
    const int kt0 = bTok0 + (kj0 < SEQ - 1 ? kj0 : SEQ - 1);  // clamp (masked anyway)
    const int kt1 = bTok0 + (kj1 < SEQ - 1 ? kj1 : SEQ - 1);
    v16bf kb[4];
    kb[0] = ldb32(Kb + (size_t)kt0 * HD + half * 16);
    kb[1] = ldb32(Kb + (size_t)kt1 * HD + half * 16);
    kb[2] = ldb32(Kb + (size_t)kt0 * HD + 32 + half * 16);
    kb[3] = ldb32(Kb + (size_t)kt1 * HD + 32 + half * 16);
    v8f S0 = (v8f)0.f, S1 = (v8f)0.f;
    S0 = wmma_bf16(qa[0], kb[0], S0);
    S1 = wmma_bf16(qa[0], kb[1], S1);
    S0 = wmma_bf16(qa[1], kb[2], S0);
    S1 = wmma_bf16(qa[1], kb[3], S1);

    // ---- V fragments: issue loads now, softmax VALU hides their latency
    int js = j0 + half * 16;
    if (js > SEQ - 16) js = SEQ - 16;     // clamp; clamped cols have p == 0
    v16bf vb[4];
#pragma unroll
    for (int nt = 0; nt < 4; ++nt)
      vb[nt] = ldb32(VT + (size_t)(nt * 16 + ln) * NTOK + bTok0 + js);

    // ---- online softmax (rows = v + 8*half, cols = lane&15 per tile)
    float p0[8], p1[8], alpha[8];
#pragma unroll
    for (int v = 0; v < 8; ++v) {
      const int qi = q0 + v + 8 * half;
      float s0 = (j0 + ln      <= qi) ? S0[v] * 0.125f : -1e30f;
      float s1 = (j0 + 16 + ln <= qi) ? S1[v] * 0.125f : -1e30f;
      float t = fmaxf(s0, s1);
      t = fmaxf(t, __shfl_xor(t, 1));
      t = fmaxf(t, __shfl_xor(t, 2));
      t = fmaxf(t, __shfl_xor(t, 4));
      t = fmaxf(t, __shfl_xor(t, 8));      // row-max within 16-lane half
      const float mn = fmaxf(m[v], t);
      const float a  = __expf(m[v] - mn);
      const float e0 = __expf(s0 - mn);
      const float e1 = __expf(s1 - mn);
      float u = e0 + e1;
      u += __shfl_xor(u, 1); u += __shfl_xor(u, 2);
      u += __shfl_xor(u, 4); u += __shfl_xor(u, 8);
      l[v] = l[v] * a + u;
      m[v] = mn;
      alpha[v] = a; p0[v] = e0; p1[v] = e1;
    }
#pragma unroll
    for (int nt = 0; nt < 4; ++nt)
#pragma unroll
      for (int v = 0; v < 8; ++v) O[nt][v] *= alpha[v];

    // ---- C-layout P -> LDS (row-major 16x32 bf16) -> A-fragment
#pragma unroll
    for (int v = 0; v < 8; ++v) {
      const int r = v + 8 * half;
      Pt[r * 32 + ln]      = f2bf(p0[v]);
      Pt[r * 32 + 16 + ln] = f2bf(p1[v]);
    }
    asm volatile("s_wait_dscnt 0x0" ::: "memory");   // in-wave LDS RAW
    const unsigned short* pp = &Pt[ln * 32 + half * 16];
    v16bf pa = mk16(*(const v4u*)pp, *(const v4u*)(pp + 8));

    // ---- O += P V
#pragma unroll
    for (int nt = 0; nt < 4; ++nt)
      O[nt] = wmma_bf16(pa, vb[nt], O[nt]);
  }

  // ---- epilogue: out = O / l (f32)
#pragma unroll
  for (int nt = 0; nt < 4; ++nt)
#pragma unroll
    for (int v = 0; v < 8; ++v) {
      const int tok = tokQ0 + v + 8 * half;
      out[(size_t)tok * HD + nt * 16 + ln] = O[nt][v] / l[v];
    }
}

// ---------------------------------------------------------------------------
extern "C" void kernel_launch(void* const* d_in, const int* in_sizes, int n_in,
                              void* d_out, int out_size, void* d_ws, size_t ws_size,
                              hipStream_t stream) {
  const float* X  = (const float*)d_in[0];
  const float* Wq = (const float*)d_in[1];
  const float* bq = (const float*)d_in[2];
  const float* Wk = (const float*)d_in[3];
  const float* bk = (const float*)d_in[4];
  const float* Wv = (const float*)d_in[5];
  const float* bv = (const float*)d_in[6];
  float* out = (float*)d_out;

  char* ws = (char*)d_ws;
  unsigned short* Wpack = (unsigned short*)(ws);
  unsigned short* Qb    = (unsigned short*)(ws + 294912);
  unsigned short* Kb    = (unsigned short*)(ws + 294912 + 2097152);
  unsigned short* VT    = (unsigned short*)(ws + 294912 + 2 * 2097152);
  // total workspace use: ~6.6 MB

  pack_weights<<<288, 32, 0, stream>>>(Wq, Wk, Wv, Wpack);
  qkv_kernel<<<128, 256, 0, stream>>>(X, Wpack, bq, bk, bv, Qb, Kb, VT);
  flash_kernel<<<1024, 32, 0, stream>>>(Qb, Kb, VT, out);
}
